// RGCNGateEncoder_37357625541114
// MI455X (gfx1250) — matrix operand order repaired and checked by reading, not compiled, not run on hardware
//
#include <hip/hip_runtime.h>
#include <math.h>

#define NREL 6
#define NBASES 30
#define D 256   // hidden size (layouts hard-coded to 256)

typedef __bf16 v8bf  __attribute__((ext_vector_type(8)));
typedef __bf16 v16bf __attribute__((ext_vector_type(16)));
typedef float  v8f   __attribute__((ext_vector_type(8)));
typedef unsigned short us8 __attribute__((ext_vector_type(8)));

static __device__ __forceinline__ unsigned short f2bf_bits(float f) {
  unsigned u = __builtin_bit_cast(unsigned, f);
  u = u + 0x7FFFu + ((u >> 16) & 1u);          // round-to-nearest-even
  return (unsigned short)(u >> 16);
}
static __device__ __forceinline__ float bf2f_bits(unsigned short s) {
  unsigned u = ((unsigned)s) << 16;
  return __builtin_bit_cast(float, u);
}

// ---------------------------------------------------------------------------
// W_r = sum_b att[r,b] * basis[b]  (r<6), W_6 = root; stored TRANSPOSED bf16:
// Wt[r][o][i] = W[r][i][o]  so WMMA B-fragments load contiguously in K.
// ---------------------------------------------------------------------------
__global__ void build_w_kernel(const float* __restrict__ att,
                               const float* __restrict__ basis,
                               const float* __restrict__ root,
                               unsigned short* __restrict__ Wt) {
  int tid = blockIdx.x * blockDim.x + threadIdx.x;
  if (tid >= 7 * D * D) return;
  int r  = tid >> 16;
  int io = tid & 65535;           // i*256 + o  (coalesced over o)
  int i = io >> 8, o = io & 255;
  float v;
  if (r < NREL) {
    v = 0.f;
#pragma unroll 6
    for (int b = 0; b < NBASES; ++b)
      v = fmaf(att[r * NBASES + b], basis[(size_t)b * (D * D) + io], v);
  } else {
    v = root[io];
  }
  Wt[((size_t)(r * D + o)) * D + i] = f2bf_bits(v);
}

// f32 -> bf16 (4 elements / thread)
__global__ void cvt_bf16_kernel(const float* __restrict__ in,
                                unsigned short* __restrict__ out, int n4) {
  int t = blockIdx.x * blockDim.x + threadIdx.x;
  if (t >= n4) return;
  float4 v = ((const float4*)in)[t];
  ushort4 o;
  o.x = f2bf_bits(v.x); o.y = f2bf_bits(v.y);
  o.z = f2bf_bits(v.z); o.w = f2bf_bits(v.w);
  ((ushort4*)out)[t] = o;
}

// ---------------------------------------------------------------------------
// xw[r] = X(bf16) @ W_r via v_wmma_f32_16x16x32_bf16.
// One wave computes a 16-row x (2 x 16-col) tile, K-loop = 8 WMMAs per tile.
// r<6: write bf16 into xw (L2-resident: 153.6MB < 192MB). r==6: f32 rootout.
// ---------------------------------------------------------------------------
__global__ void __launch_bounds__(256)
gemm_wmma_kernel(const unsigned short* __restrict__ xbf,
                 const unsigned short* __restrict__ Wt,
                 unsigned short* __restrict__ xw,
                 float* __restrict__ rootout, int nnodes) {
  const int lane  = threadIdx.x & 31;
  const int wave  = threadIdx.x >> 5;
  const int mtile = blockIdx.x * 8 + wave;
  const int mtiles = nnodes >> 4;
  if (mtile >= mtiles) return;                 // wave-uniform: EXEC stays all-1s
  const int r   = blockIdx.z;
  const int n0a = blockIdx.y * 16;             // col tiles y and y+8
  const int n0b = n0a + 128;
  const int lr  = lane & 15;
  const int hi  = lane >> 4;

  // A fragment: lane holds row (mtile*16+lr); lo-lanes K {k0..k0+7, k0+16..23},
  // hi-lanes K {k0+8..15, k0+24..31}  (per CDNA5 16-bit A layout)
  const unsigned short* arow  = xbf + ((size_t)(mtile * 16 + lr)) * D + (hi ? 8 : 0);
  // B fragment: lane holds column n; same K layout as A (lo: k0..7, k0+16..23; hi: k0+8..15, k0+24..31)
  const unsigned short* brow0 = Wt + ((size_t)(r * D + n0a + lr)) * D + (hi ? 8 : 0);
  const unsigned short* brow1 = Wt + ((size_t)(r * D + n0b + lr)) * D + (hi ? 8 : 0);

  v8f acc0 = {}; v8f acc1 = {};
#pragma unroll
  for (int kk = 0; kk < 8; ++kk) {
    const int k0 = kk * 32;
    v8bf alo = *(const v8bf*)(const void*)(arow + k0);
    v8bf ahi = *(const v8bf*)(const void*)(arow + k0 + 16);
    v16bf a = __builtin_shufflevector(alo, ahi, 0,1,2,3,4,5,6,7,8,9,10,11,12,13,14,15);

    v8bf b0l = *(const v8bf*)(const void*)(brow0 + k0);
    v8bf b0h = *(const v8bf*)(const void*)(brow0 + k0 + 16);
    v16bf b0 = __builtin_shufflevector(b0l, b0h, 0,1,2,3,4,5,6,7,8,9,10,11,12,13,14,15);
    acc0 = __builtin_amdgcn_wmma_f32_16x16x32_bf16(false, a, false, b0,
                                                   (short)0, acc0, false, false);
    v8bf b1l = *(const v8bf*)(const void*)(brow1 + k0);
    v8bf b1h = *(const v8bf*)(const void*)(brow1 + k0 + 16);
    v16bf b1 = __builtin_shufflevector(b1l, b1h, 0,1,2,3,4,5,6,7,8,9,10,11,12,13,14,15);
    acc1 = __builtin_amdgcn_wmma_f32_16x16x32_bf16(false, a, false, b1,
                                                   (short)0, acc1, false, false);
  }

  // C/D layout: element j -> row m0 + j + hi*8, col n0 + (lane&15)
  const int colA = n0a + lr, colB = n0b + lr;
#pragma unroll
  for (int j = 0; j < 8; ++j) {
    int row = mtile * 16 + j + hi * 8;
    if (r < NREL) {
      size_t base = ((size_t)r * nnodes + row) * D;
      xw[base + colA] = f2bf_bits(acc0[j]);
      xw[base + colB] = f2bf_bits(acc1[j]);
    } else {
      size_t base = (size_t)row * D;
      rootout[base + colA] = acc0[j];
      rootout[base + colB] = acc1[j];
    }
  }
}

// gdot[r,n] = xw[r,n,:] . gate_w  (one wave per row; 8 bf16 per lane)
__global__ void __launch_bounds__(256)
gdot_kernel(const unsigned short* __restrict__ xw,
            const float* __restrict__ gate_w,
            float* __restrict__ gdot, int nnodes) {
  int w = (blockIdx.x * blockDim.x + threadIdx.x) >> 5;
  int lane = threadIdx.x & 31;
  if (w >= NREL * nnodes) return;
  const unsigned short* row = xw + (size_t)w * D + lane * 8;
  us8 m = *(const us8*)(const void*)row;
  float4 g0 = ((const float4*)gate_w)[lane * 2];
  float4 g1 = ((const float4*)gate_w)[lane * 2 + 1];
  float s = bf2f_bits(m[0]) * g0.x + bf2f_bits(m[1]) * g0.y +
            bf2f_bits(m[2]) * g0.z + bf2f_bits(m[3]) * g0.w +
            bf2f_bits(m[4]) * g1.x + bf2f_bits(m[5]) * g1.y +
            bf2f_bits(m[6]) * g1.z + bf2f_bits(m[7]) * g1.w;
#pragma unroll
  for (int off = 16; off > 0; off >>= 1) s += __shfl_xor(s, off, 32);
  if (lane == 0) gdot[w] = s;
}

__global__ void deg_kernel(const int* __restrict__ dst,
                           float* __restrict__ deg, int E) {
  int e = blockIdx.x * blockDim.x + threadIdx.x;
  if (e < E) atomicAdd(&deg[dst[e]], 1.0f);
}

// One wave per edge: scalar gate from gdot, gather bf16 row (L2), f32 atomic
// scatter into agg (51MB, L2-resident).
__global__ void __launch_bounds__(256)
edge_kernel(const unsigned short* __restrict__ xw,
            const float* __restrict__ gdot,
            const float* __restrict__ gate_b,
            const int* __restrict__ src, const int* __restrict__ dst,
            const int* __restrict__ et,
            float* __restrict__ agg, int nnodes, int E) {
  int e = (blockIdx.x * blockDim.x + threadIdx.x) >> 5;
  int lane = threadIdx.x & 31;
  if (e >= E) return;
  int s = src[e], d = dst[e], r = et[e];
  float z = gdot[r * nnodes + s] + gate_b[0];
  float g = 1.0f / (1.0f + __expf(-z));
  const unsigned short* row = xw + ((size_t)r * nnodes + s) * D + lane * 8;
  us8 m = *(const us8*)(const void*)row;
  float* out = agg + (size_t)d * D + lane * 8;
#pragma unroll
  for (int i = 0; i < 8; ++i) atomicAdd(out + i, g * bf2f_bits(m[i]));
}

// out = agg/max(deg,1) + x@root + bias ; optional relu; writes f32 and/or bf16
__global__ void finalize_kernel(const float* __restrict__ agg,
                                const float* __restrict__ deg,
                                const float* __restrict__ rootout,
                                const float* __restrict__ bias,
                                float* __restrict__ outf,
                                unsigned short* __restrict__ outbf,
                                int nnodes, int do_relu) {
  int i = blockIdx.x * blockDim.x + threadIdx.x;
  if (i >= nnodes * D) return;
  int n = i >> 8, c = i & 255;
  float dg = deg[n]; dg = dg > 1.f ? dg : 1.f;
  float v = agg[i] / dg + rootout[i] + bias[c];
  if (do_relu) v = v > 0.f ? v : 0.f;
  if (outf)  outf[i] = v;
  if (outbf) outbf[i] = f2bf_bits(v);
}

extern "C" void kernel_launch(void* const* d_in, const int* in_sizes, int n_in,
                              void* d_out, int out_size, void* d_ws, size_t ws_size,
                              hipStream_t stream) {
  const float* x      = (const float*)d_in[0];
  const int*   esrc   = (const int*)d_in[1];
  const int*   edst   = (const int*)d_in[2];
  const int*   etyp   = (const int*)d_in[3];
  const float* basis1 = (const float*)d_in[4];
  const float* att1   = (const float*)d_in[5];
  const float* root1  = (const float*)d_in[6];
  const float* bias1  = (const float*)d_in[7];
  const float* gw1    = (const float*)d_in[8];
  const float* gb1    = (const float*)d_in[9];
  const float* basis2 = (const float*)d_in[10];
  const float* att2   = (const float*)d_in[11];
  const float* root2  = (const float*)d_in[12];
  const float* bias2  = (const float*)d_in[13];
  const float* gw2    = (const float*)d_in[14];
  const float* gb2    = (const float*)d_in[15];

  const int N = in_sizes[0] / D;   // 50000
  const int E = in_sizes[1];       // 800000

  // Workspace carve-up (≈284 MB total)
  char* p = (char*)d_ws;
  auto take = [&](size_t bytes) -> char* {
    char* q = p; p += (bytes + 255) & ~(size_t)255; return q;
  };
  unsigned short* Wt      = (unsigned short*)take((size_t)7 * D * D * 2);
  unsigned short* xbf     = (unsigned short*)take((size_t)N * D * 2);
  unsigned short* xw      = (unsigned short*)take((size_t)NREL * N * D * 2);
  float*          rootout = (float*)take((size_t)N * D * 4);
  float*          gdotb   = (float*)take((size_t)NREL * N * 4);
  float*          agg     = (float*)take((size_t)N * D * 4);
  float*          degb    = (float*)take((size_t)N * 4);

  // Degree (shared by both layers: same edge list)
  hipMemsetAsync(degb, 0, (size_t)N * 4, stream);
  deg_kernel<<<(E + 255) / 256, 256, 0, stream>>>(edst, degb, E);

  // x -> bf16
  int n4 = N * (D / 4);
  cvt_bf16_kernel<<<(n4 + 255) / 256, 256, 0, stream>>>(x, xbf, n4);

  auto run_layer = [&](const float* basis, const float* att, const float* root,
                       const float* bias, const float* gw, const float* gb,
                       float* outf, unsigned short* outbf, int relu) {
    build_w_kernel<<<(7 * D * D + 255) / 256, 256, 0, stream>>>(att, basis, root, Wt);
    dim3 gg((N / 16 + 7) / 8, 8, 7);
    gemm_wmma_kernel<<<gg, 256, 0, stream>>>(xbf, Wt, xw, rootout, N);
    gdot_kernel<<<(NREL * N * 32 + 255) / 256, 256, 0, stream>>>(xw, gw, gdotb, N);
    hipMemsetAsync(agg, 0, (size_t)N * D * 4, stream);
    edge_kernel<<<(E * 32 + 255) / 256, 256, 0, stream>>>(xw, gdotb, gb, esrc,
                                                          edst, etyp, agg, N, E);
    finalize_kernel<<<(N * D + 255) / 256, 256, 0, stream>>>(agg, degb, rootout,
                                                             bias, outf, outbf,
                                                             N, relu);
  };

  // Layer 1: relu, output straight to bf16 GEMM-input buffer for layer 2
  run_layer(basis1, att1, root1, bias1, gw1, gb1, nullptr, xbf, 1);
  // Layer 2: f32 output
  run_layer(basis2, att2, root2, bias2, gw2, gb2, (float*)d_out, nullptr, 0);
}
